// KMeans_9294309229230
// MI455X (gfx1250) — matrix operand order repaired and checked by reading, not compiled, not run on hardware
//
#include <hip/hip_runtime.h>

// ---------------------------------------------------------------------------
// K-means single step for MI455X (gfx1250, wave32, WMMA, async-to-LDS).
//   argmin_k ||x-c_k||^2  ==  argmin_k (||c_k||^2 - 2 x.c_k)
//   GEMM via v_wmma_f32_16x16x32_f16 (double-buffered B frags from LDS),
//   x tiles prefetched with global_load_async_to_lds_b128 (ASYNCcnt),
//   counts/sums accumulated in LDS, flushed once per WG with f32 atomics.
// ---------------------------------------------------------------------------

typedef __attribute__((ext_vector_type(16))) _Float16 v16h;
typedef __attribute__((ext_vector_type(8)))  _Float16 v8h;
typedef __attribute__((ext_vector_type(4)))  _Float16 v4h;
typedef __attribute__((ext_vector_type(8)))  float    v8f;
typedef __attribute__((ext_vector_type(4)))  float    v4f;

#define NROWS 524288
#define DDIM  64
#define KCL   512
#define WGSZ  256
#define GRID  256
#define ROWS_PER_WG (NROWS / GRID)        // 2048
#define TILES_PER_WG (ROWS_PER_WG / 16)   // 128
#define ITERS_PER_WAVE (TILES_PER_WG / 8) // 16
#define CP 72     // f16 centers LDS pitch (breaks stride-64 bank conflicts)
#define XP 68     // f32 x-tile LDS pitch (row stride 68*4 B, bank-clean)

union AFrag { v16h v; v4h q[4]; };
union BFrag { v16h v; v8h d[2]; };

// LDS carve-up (bytes):
//   sums   KCL*DDIM*4          = 131072
//   xtile  2*8*16*XP*4         =  69632   (double buffer, per wave)
//   c2     KCL*4               =   2048
//   cnt    KCL*4               =   2048
//   assign 128*4               =    512
//   cf16   KCL*CP*2            =  73728
#define SMEM_BYTES (KCL*DDIM*4 + 2*8*16*XP*4 + KCL*4 + KCL*4 + 128*4 + KCL*CP*2)

__device__ __forceinline__ void async_b128(unsigned lds_byte_off, const float* g) {
  // GLOBAL_LOAD_ASYNC_TO_LDS_B128: VDST = per-lane LDS byte address,
  // VADDR = per-lane 64-bit global address. Tracked by ASYNCcnt.
  asm volatile("global_load_async_to_lds_b128 %0, %1, off"
               :: "v"(lds_byte_off), "v"(g) : "memory");
}

__global__ __launch_bounds__(WGSZ) void kmeans_main(
    const float* __restrict__ x, const float* __restrict__ centers,
    float* __restrict__ gsums, float* __restrict__ gcounts)
{
  extern __shared__ __align__(16) char smem[];
  float*    lds_sums = (float*)smem;                       // [KCL][DDIM]
  float*    lds_x    = lds_sums + KCL * DDIM;              // [2][8][16][XP]
  float*    lds_c2   = lds_x + 2 * 8 * 16 * XP;            // [KCL]
  float*    lds_cnt  = lds_c2 + KCL;                       // [KCL]
  int*      lds_asn  = (int*)(lds_cnt + KCL);              // [8][16]
  _Float16* cf16     = (_Float16*)(lds_asn + 128);         // [KCL][CP]

  const int tid   = threadIdx.x;
  const int lane  = tid & 31;
  const int wv    = tid >> 5;          // wave id 0..7
  const int mrow  = lane & 15;         // row within 16-row tile (A), col (B/D)
  const int hh    = lane >> 4;         // half-wave id
  const int koffA = hh * 8;            // A-fragment K base per ISA layout
  const int koffB = hh * 8;            // B-fragment K base per ISA layout
  const int wgBase = blockIdx.x * ROWS_PER_WG;

  // ---- Stage 0: centers -> f16 LDS (+ ||c||^2), zero accumulators --------
  for (int k = tid; k < KCL; k += WGSZ) {
    const float* cr = centers + k * DDIM;
    _Float16*    ch = cf16 + k * CP;
    float s = 0.0f;
#pragma unroll
    for (int j = 0; j < DDIM / 4; ++j) {
      v4f c = *(const v4f*)(cr + j * 4);
      s += c.x * c.x + c.y * c.y + c.z * c.z + c.w * c.w;
      *(v4h*)(ch + j * 4) = __builtin_convertvector(c, v4h);
    }
    lds_c2[k] = s;
  }
  for (int i = tid; i < KCL * DDIM; i += WGSZ) lds_sums[i] = 0.0f;
  for (int i = tid; i < KCL; i += WGSZ)        lds_cnt[i]  = 0.0f;
  __syncthreads();

  // Per-wave x-tile buffers (LDS byte offsets for the async engine).
  float* xbuf0 = lds_x + (0 * 8 + wv) * 16 * XP;
  float* xbuf1 = lds_x + (1 * 8 + wv) * 16 * XP;
  const unsigned xboff[2] = { (unsigned)(size_t)(void*)xbuf0,
                              (unsigned)(size_t)(void*)xbuf1 };

  // Issue one 16x64 f32 tile copy: 256 b128 chunks = 8 per lane.
  auto issue_tile = [&](int buf, int t) {
    const int rowBase = wgBase + t * 16;
#pragma unroll
    for (int j = 0; j < 8; ++j) {
      const int linear = j * 32 + lane;
      const int row = linear >> 4;
      const int c4  = (linear & 15) * 4;
      const float* g = x + (size_t)(rowBase + row) * DDIM + c4;
      const unsigned loff = xboff[buf] + (unsigned)((row * XP + c4) * 4);
      async_b128(loff, g);
    }
  };

  issue_tile(0, wv);  // prime the pipeline

  for (int it = 0; it < ITERS_PER_WAVE; ++it) {
    const int t   = wv + it * 8;
    const int cur = it & 1;

    // Prefetch next tile into the other buffer (WAR fenced by dscnt wait).
    asm volatile("s_wait_dscnt 0x0" ::: "memory");
    if (it + 1 < ITERS_PER_WAVE) issue_tile(cur ^ 1, t + 8);
    // Wait for the previous copy (allow the 8 just-issued to stay in flight).
    asm volatile("s_wait_asynccnt 0x8" ::: "memory");

    // ---- Build A fragments (f32 LDS -> f16 regs) per ISA 16-bit A layout.
    const float* xb = lds_x + ((cur * 8 + wv) * 16 + mrow) * XP;
    AFrag a0, a1;
#pragma unroll
    for (int j = 0; j < 4; ++j) {
      const int dbase = j * 16 + koffA;
      v4f f0 = *(const v4f*)(xb + dbase);
      v4f f1 = *(const v4f*)(xb + dbase + 4);
      AFrag& af = (j < 2) ? a0 : a1;
      af.q[(j & 1) * 2]     = __builtin_convertvector(f0, v4h);
      af.q[(j & 1) * 2 + 1] = __builtin_convertvector(f1, v4h);
    }

    float minv[8];
    int   mini[8];
#pragma unroll
    for (int r = 0; r < 8; ++r) { minv[r] = __builtin_inff(); mini[r] = 0; }

    // ---- 32 cluster tiles, software-pipelined B fragments ----------------
    auto loadB = [&](int nt, BFrag& b0, BFrag& b1, float& c2v) {
      const int cl = nt * 16 + mrow;
      const _Float16* crow = cf16 + cl * CP + koffB;
      b0.d[0] = *(const v8h*)(crow);
      b0.d[1] = *(const v8h*)(crow + 16);
      b1.d[0] = *(const v8h*)(crow + 32);
      b1.d[1] = *(const v8h*)(crow + 48);
      c2v = lds_c2[cl];
    };
    auto minUpd = [&](const v8f& acc, float c2v, int cl) {
#pragma unroll
      for (int r = 0; r < 8; ++r) {
        float dv = fmaf(-2.0f, acc[r], c2v);
        if (dv < minv[r]) { minv[r] = dv; mini[r] = cl; }
      }
    };

    BFrag b0a, b1a, b0b, b1b;
    float c2a, c2b;
    loadB(0, b0a, b1a, c2a);
#pragma unroll 2
    for (int nt = 0; nt < KCL / 16; nt += 2) {
      loadB(nt + 1, b0b, b1b, c2b);          // in flight during first WMMA pair
      v8f acc = {};
      acc = __builtin_amdgcn_wmma_f32_16x16x32_f16(false, a0.v, false, b0a.v,
                                                   (short)0, acc, false, false);
      acc = __builtin_amdgcn_wmma_f32_16x16x32_f16(false, a1.v, false, b1a.v,
                                                   (short)0, acc, false, false);
      minUpd(acc, c2a, nt * 16 + mrow);
      if (nt + 2 < KCL / 16) loadB(nt + 2, b0a, b1a, c2a);
      v8f acc2 = {};
      acc2 = __builtin_amdgcn_wmma_f32_16x16x32_f16(false, a0.v, false, b0b.v,
                                                    (short)0, acc2, false, false);
      acc2 = __builtin_amdgcn_wmma_f32_16x16x32_f16(false, a1.v, false, b1b.v,
                                                    (short)0, acc2, false, false);
      minUpd(acc2, c2b, (nt + 1) * 16 + mrow);
    }

    // ---- Cross-lane argmin within each 16-lane half (D: N = lane&15) -----
#pragma unroll
    for (int off = 1; off < 16; off <<= 1) {
#pragma unroll
      for (int r = 0; r < 8; ++r) {
        float ov = __shfl_xor(minv[r], off, 32);
        int   oi = __shfl_xor(mini[r], off, 32);
        bool take = (ov < minv[r]) | ((ov == minv[r]) & (oi < mini[r]));
        minv[r] = take ? ov : minv[r];
        mini[r] = take ? oi : mini[r];
      }
    }

    // Lanes 0 / 16 publish assignments (rows 0-7 / 8-15) + bump counts.
    if (mrow == 0) {
#pragma unroll
      for (int r = 0; r < 8; ++r) {
        lds_asn[wv * 16 + hh * 8 + r] = mini[r];
        atomicAdd(&lds_cnt[mini[r]], 1.0f);
      }
    }
    asm volatile("s_wait_dscnt 0x0" ::: "memory");  // intra-wave LDS RAW

    // ---- Cooperative scatter of the f32 x-tile into per-cluster sums -----
    const float* xsrc = lds_x + (cur * 8 + wv) * 16 * XP;
    for (int r = 0; r < 16; ++r) {
      const int    a   = lds_asn[wv * 16 + r];
      float*       dst = lds_sums + a * DDIM;
      const float* src = xsrc + r * XP;
      atomicAdd(&dst[lane * 2],     src[lane * 2]);
      atomicAdd(&dst[lane * 2 + 1], src[lane * 2 + 1]);
    }
  }

  // ---- Flush workgroup-local accumulators to global (f32 atomics) --------
  __syncthreads();
  for (int i = tid; i < KCL * DDIM; i += WGSZ) atomicAdd(&gsums[i], lds_sums[i]);
  for (int i = tid; i < KCL; i += WGSZ)        atomicAdd(&gcounts[i], lds_cnt[i]);
}

__global__ void kmeans_zero_ws(float* __restrict__ ws, int n) {
  int i = blockIdx.x * blockDim.x + threadIdx.x;
  if (i < n) ws[i] = 0.0f;
}

__global__ void kmeans_finalize(const float* __restrict__ gsums,
                                const float* __restrict__ gcounts,
                                const float* __restrict__ centers,
                                float* __restrict__ out) {
  int i = blockIdx.x * blockDim.x + threadIdx.x;
  if (i < KCL * DDIM) {
    int   k = i / DDIM;
    float c = gcounts[k];
    float m = gsums[i] / fmaxf(c, 1.0f);
    out[i] = (c > 0.0f) ? m : centers[i];
  } else if (i < KCL * DDIM + KCL) {
    out[i] = gcounts[i - KCL * DDIM];
  }
}

extern "C" void kernel_launch(void* const* d_in, const int* in_sizes, int n_in,
                              void* d_out, int out_size, void* d_ws, size_t ws_size,
                              hipStream_t stream) {
  (void)in_sizes; (void)n_in; (void)out_size; (void)ws_size;
  const float* x       = (const float*)d_in[0];   // [N, 64] f32
  const float* centers = (const float*)d_in[1];   // [512, 64] f32
  float* gsums   = (float*)d_ws;                  // [512*64]
  float* gcounts = gsums + KCL * DDIM;            // [512]
  float* out     = (float*)d_out;                 // [512*64 + 512]

  const int nacc = KCL * DDIM + KCL;              // 33280
  kmeans_zero_ws<<<(nacc + 255) / 256, 256, 0, stream>>>(gsums, nacc);
  kmeans_main<<<GRID, WGSZ, SMEM_BYTES, stream>>>(x, centers, gsums, gcounts);
  kmeans_finalize<<<(nacc + 255) / 256, 256, 0, stream>>>(gsums, gcounts, centers, out);
}
